// LSTM_38491496907304
// MI455X (gfx1250) — matrix operand order repaired and checked by reading, not compiled, not run on hardware
//
#include <hip/hip_runtime.h>

// ---------------------------------------------------------------------------
// LSTM forward for MI455X (gfx1250), bf16 WMMA (v_wmma_f32_16x16x32_bf16).
// B=128, T=512, IN=H=1024, L=2, OUT=20.
// Weights (bf16, 33.5MB) stay L2-resident (192MB L2) -> compute-bound GEMMs.
// K=2048 reduction split across 2 waves per tile-group (LDS combine) to
// double exposed parallelism per sequential step (1024 waves/step-layer).
// ---------------------------------------------------------------------------

typedef __attribute__((ext_vector_type(16))) __bf16 v16bf;
typedef __attribute__((ext_vector_type(8)))  float  v8f;
typedef __attribute__((ext_vector_type(4)))  int    i4;
typedef __attribute__((ext_vector_type(2)))  float  f2;

union Frag { i4 q[2]; v16bf v; };

#define BB   128
#define TT   512
#define HH   1024
#define G4   4096          // 4*H
#define LDA  1028          // LDS row stride in dwords (1024 + 4 pad, kills bank conflicts)

__device__ __forceinline__ unsigned short f2bf(float f) {
    unsigned int u = __float_as_uint(f);
    unsigned int r = (u + 0x7FFFu + ((u >> 16) & 1u)) >> 16;   // round-to-nearest-even
    return (unsigned short)r;
}
__device__ __forceinline__ float sigf(float x)     { return 1.f / (1.f + __expf(-x)); }
__device__ __forceinline__ float tanhfast(float x) { return 1.f - 2.f / (1.f + __expf(2.f * x)); }

// ---------------------------------------------------------------------------
// Pack Wx|Wh (fp32, [L][4096][1024] each) into bf16 WMMA B-fragment order:
// wf[l][ntile(256)][kstep(64)][lane(32)][half(16)]
//   lane 0-15 : N = ntile*16 + lane,      K = kstep*32 + (h&7) + 16*(h>>3)   (h = 0..15)
//   lane 16-31: N = ntile*16 + lane - 16, K = kstep*32 + 8 + (h&7) + 16*(h>>3)
// K < 1024 -> Wx, else Wh.
// ---------------------------------------------------------------------------
__global__ void pack_w(const float* __restrict__ Wx, const float* __restrict__ Wh,
                       unsigned short* __restrict__ wf) {
    size_t base = ((size_t)blockIdx.x * 256 + threadIdx.x) * 4;
#pragma unroll
    for (int i = 0; i < 4; ++i) {
        size_t e   = base + i;                 // e < 2^24
        int h      = (int)(e & 15);
        int lane   = (int)((e >> 4)  & 31);
        int ks     = (int)((e >> 9)  & 63);
        int ntile  = (int)((e >> 15) & 255);
        int l      = (int)(e >> 23);
        int n      = ntile * 16 + (lane & 15);
        int K      = ks * 32 + ((h >> 3) << 4) + ((lane >> 4) << 3) + (h & 7);
        float v    = (K < HH) ? Wx[((size_t)l * G4 + n) * HH + K]
                              : Wh[((size_t)l * G4 + n) * HH + (K - HH)];
        wf[e] = f2bf(v);
    }
}

__global__ void pack_bias(const float* __restrict__ bx, const float* __restrict__ bh,
                          float* __restrict__ bias) {
    int i = blockIdx.x * 256 + threadIdx.x;
    if (i < 2 * G4) bias[i] = bx[i] + bh[i];
}

__global__ void zero_h(unsigned int* __restrict__ p) {
    size_t i = ((size_t)blockIdx.x * 256 + threadIdx.x) * 4;   // 524288 dwords total
    i4 z = 0;
    *(i4*)(p + i) = z;
}

// ---------------------------------------------------------------------------
// One step of one layer, fused: gates = A[128,2048] * W'[2048,4096] + bias,
// A = [inp | h_in] (bf16), then LSTM cell elementwise, write hy (bf16 + opt f32).
// Grid: 128 blocks = 8 m-tiles x 16 n-blocks; 8 waves/block = 4 gate-tile
// groups x 2 K-split halves. Waves 4-7 (K half 1) dump partials to LDS,
// waves 0-3 combine + run the register-resident LSTM cell epilogue.
// ---------------------------------------------------------------------------
__global__ void __launch_bounds__(256)
lstm_step(const float* __restrict__ x_f32,              // layer0: x base, else null
          const unsigned short* __restrict__ inp_bf,    // layer1: hy0 (bf16), else null
          const unsigned short* __restrict__ h_in,      // [128][1024] bf16
          unsigned short* __restrict__ h_out,           // [128][1024] bf16
          const unsigned short* __restrict__ wfrag,     // packed bf16 B-fragments
          const float* __restrict__ bias,               // [4096] (bx+bh)
          float* __restrict__ htop,                     // layer1: fp32 h, else null
          int t) {
    __shared__ unsigned int Ash[16 * LDA];               // 64.25 KB A panel
    __shared__ float Red[16 * 256];                      // 16 KB K-split partials

    const int tid   = threadIdx.x;
    const int wave  = tid >> 5;
    const int lane  = tid & 31;
    const int mtile = blockIdx.x >> 4;                   // 0..7  (16 batch rows)
    const int nblk  = blockIdx.x & 15;                   // 0..15 (4 gate-col groups)
    const int grp   = wave & 3;                          // gate-tile group in block
    const int khalf = wave >> 2;                         // 0: K 0..1023, 1: K 1024..2047
    const int ctile = nblk * 4 + grp;                    // 0..63 (16 H-columns each)

    // ---- stage A panel (16 rows x 2048 halves) into LDS -------------------
    if (x_f32) {
#pragma unroll 4
        for (int d = 0; d < 64; ++d) {
            int g   = tid + d * 256;                     // 0..16383 dwords
            int row = g >> 10;
            int kd  = g & 1023;
            int b   = mtile * 16 + row;
            unsigned int val;
            if (kd < 512) {
                f2 xv = *(const f2*)(x_f32 + ((size_t)b * TT + t) * HH + kd * 2);
                val = (unsigned int)f2bf(xv.x) | ((unsigned int)f2bf(xv.y) << 16);
            } else {
                val = ((const unsigned int*)h_in)[b * 512 + (kd - 512)];
            }
            Ash[row * LDA + kd] = val;
        }
    } else {
#pragma unroll 4
        for (int d = 0; d < 64; ++d) {
            int g   = tid + d * 256;
            int row = g >> 10;
            int kd  = g & 1023;
            int b   = mtile * 16 + row;
            unsigned int val = (kd < 512)
                ? ((const unsigned int*)inp_bf)[b * 512 + kd]
                : ((const unsigned int*)h_in)[b * 512 + (kd - 512)];
            Ash[row * LDA + kd] = val;
        }
    }
    __syncthreads();

    // ---- main GEMM loop: 32 k-steps of 32 per wave, 4 gate accumulators ---
    v8f acc0 = {}, acc1 = {}, acc2 = {}, acc3 = {};
    const int rowA = lane & 15;
    const int hiA  = lane >> 4;
    const int ks0  = khalf * 32;
    const unsigned int* arow = &Ash[rowA * LDA];

    const i4* wf0 = (const i4*)wfrag + ((size_t)((0 * 64 + ctile) * 64 + ks0) * 32 + lane) * 2;
    const i4* wf1 = (const i4*)wfrag + ((size_t)((1 * 64 + ctile) * 64 + ks0) * 32 + lane) * 2;
    const i4* wf2 = (const i4*)wfrag + ((size_t)((2 * 64 + ctile) * 64 + ks0) * 32 + lane) * 2;
    const i4* wf3 = (const i4*)wfrag + ((size_t)((3 * 64 + ctile) * 64 + ks0) * 32 + lane) * 2;

#pragma unroll 2
    for (int ks = 0; ks < 32; ++ks) {
        // A fragment per ISA 16-bit A 16x32 layout:
        // lanes 0-15: K+0..7 and K+16..23 ; lanes 16-31: K+8..15 and K+24..31
        Frag fa;
        const i4* pa = (const i4*)(arow + ((ks0 + ks) * 16 + hiA * 4));
        fa.q[0] = pa[0];
        fa.q[1] = pa[2];                                 // +8 dwords = +16 halves

        Frag b0, b1, b2, b3;
        const i4* p0 = wf0 + (size_t)ks * 64;            // 32 lanes * 2 i4 per kstep
        const i4* p1 = wf1 + (size_t)ks * 64;
        const i4* p2 = wf2 + (size_t)ks * 64;
        const i4* p3 = wf3 + (size_t)ks * 64;
        b0.q[0] = p0[0]; b0.q[1] = p0[1];
        b1.q[0] = p1[0]; b1.q[1] = p1[1];
        b2.q[0] = p2[0]; b2.q[1] = p2[1];
        b3.q[0] = p3[0]; b3.q[1] = p3[1];

        acc0 = __builtin_amdgcn_wmma_f32_16x16x32_bf16(false, fa.v, false, b0.v, (short)0, acc0, false, false);
        acc1 = __builtin_amdgcn_wmma_f32_16x16x32_bf16(false, fa.v, false, b1.v, (short)0, acc1, false, false);
        acc2 = __builtin_amdgcn_wmma_f32_16x16x32_bf16(false, fa.v, false, b2.v, (short)0, acc2, false, false);
        acc3 = __builtin_amdgcn_wmma_f32_16x16x32_bf16(false, fa.v, false, b3.v, (short)0, acc3, false, false);
    }

    // ---- K-split combine: waves 4-7 export partials (bank-conflict-free) --
    if (khalf) {
#pragma unroll
        for (int j = 0; j < 8; ++j) {
            Red[(grp * 4 + 0) * 256 + j * 32 + lane] = acc0[j];
            Red[(grp * 4 + 1) * 256 + j * 32 + lane] = acc1[j];
            Red[(grp * 4 + 2) * 256 + j * 32 + lane] = acc2[j];
            Red[(grp * 4 + 3) * 256 + j * 32 + lane] = acc3[j];
        }
    }
    __syncthreads();

    if (!khalf) {
        // ---- bias + partner partial + LSTM cell, write hy -----------------
        const int col = lane & 15;
        const int n   = ctile * 16 + col;
        const float bi = bias[n];
        const float bf = bias[HH + n];
        const float bg = bias[2 * HH + n];
        const float bo = bias[3 * HH + n];
        const int mbase = (lane >> 4) * 8;

#pragma unroll
        for (int j = 0; j < 8; ++j) {
            float iv = acc0[j] + Red[(grp * 4 + 0) * 256 + j * 32 + lane] + bi;
            float fv = acc1[j] + Red[(grp * 4 + 1) * 256 + j * 32 + lane] + bf;
            float gv = acc2[j] + Red[(grp * 4 + 2) * 256 + j * 32 + lane] + bg;
            float ov = acc3[j] + Red[(grp * 4 + 3) * 256 + j * 32 + lane] + bo;
            // faithful to reference bug: cy = tanh(g)*(sig(f)+sig(i)), no c carry
            float cy = tanhfast(gv) * (sigf(fv) + sigf(iv));
            float hy = tanhfast(cy) * sigf(ov);
            int b = mtile * 16 + mbase + j;
            h_out[(size_t)b * HH + n] = f2bf(hy);
            if (htop) htop[(size_t)b * HH + n] = hy;
        }
    }
}

// ---------------------------------------------------------------------------
// out[128,20] = htop @ Wout^T + bout   (tiny, fp32)
// ---------------------------------------------------------------------------
__global__ void out_kernel(const float* __restrict__ htop, const float* __restrict__ Wout,
                           const float* __restrict__ bout, float* __restrict__ out) {
    int idx = blockIdx.x * 256 + threadIdx.x;
    if (idx >= BB * 20) return;
    int b = idx / 20, o = idx % 20;
    const float* h = htop + (size_t)b * HH;
    const float* w = Wout + (size_t)o * HH;
    float s = bout[o];
#pragma unroll 8
    for (int k = 0; k < HH; ++k) s += h[k] * w[k];
    out[idx] = s;
}

// ---------------------------------------------------------------------------
extern "C" void kernel_launch(void* const* d_in, const int* in_sizes, int n_in,
                              void* d_out, int out_size, void* d_ws, size_t ws_size,
                              hipStream_t stream) {
    const float* x    = (const float*)d_in[0];
    const float* Wx   = (const float*)d_in[1];
    const float* bx   = (const float*)d_in[2];
    const float* Wh   = (const float*)d_in[3];
    const float* bh   = (const float*)d_in[4];
    const float* Wout = (const float*)d_in[5];
    const float* bout = (const float*)d_in[6];
    float* out        = (float*)d_out;

    char* ws = (char*)d_ws;
    const size_t WFRAG_L = 2048ull * 4096;                    // halves per layer (2^23)
    unsigned short* wf   = (unsigned short*)ws;               // 2 * 16 MB
    float* bias          = (float*)(ws + 2 * WFRAG_L * 2);    // 32 KB
    unsigned short* hbuf = (unsigned short*)((char*)(bias + 2 * G4));   // 4 x 256 KB
    float* htop          = (float*)((char*)(hbuf + 4ull * BB * HH));    // 512 KB
    const size_t HSZ     = (size_t)BB * HH;

    pack_w   <<<16384, 256, 0, stream>>>(Wx, Wh, wf);
    pack_bias<<<32,    256, 0, stream>>>(bx, bh, bias);
    zero_h   <<<512,   256, 0, stream>>>((unsigned int*)hbuf);

    for (int t = 0; t < TT; ++t) {
        int p = t & 1;
        unsigned short* h0i = hbuf + (size_t)(0 + p)     * HSZ;
        unsigned short* h0o = hbuf + (size_t)(0 + (1-p)) * HSZ;
        unsigned short* h1i = hbuf + (size_t)(2 + p)     * HSZ;
        unsigned short* h1o = hbuf + (size_t)(2 + (1-p)) * HSZ;
        lstm_step<<<128, 256, 0, stream>>>(x, nullptr, h0i, h0o, wf,           bias,      nullptr, t);
        lstm_step<<<128, 256, 0, stream>>>(nullptr, h0o, h1i, h1o, wf + WFRAG_L, bias + G4, htop,   t);
    }

    out_kernel<<<10, 256, 0, stream>>>(htop, Wout, bout, out);
    (void)in_sizes; (void)n_in; (void)out_size; (void)ws_size;
}
